// MultiHeadAttention_49486613184863
// MI455X (gfx1250) — matrix operand order repaired and checked by reading, not compiled, not run on hardware
//
#include <hip/hip_runtime.h>
#include <stdint.h>
#include <stddef.h>

typedef __attribute__((ext_vector_type(16))) __bf16 v16bf;
typedef __attribute__((ext_vector_type(8)))  __bf16 v8bf;
typedef __attribute__((ext_vector_type(4)))  __bf16 v4bf;
typedef __attribute__((ext_vector_type(8)))  float  v8f;
typedef __attribute__((ext_vector_type(4)))  float  v4f;
typedef __attribute__((ext_vector_type(4)))  int    v4i;

// Pointer types for the async-to-LDS builtin: global (AS1) and LDS (AS3) int4*
typedef __attribute__((address_space(1))) v4i* gv4i_p;
typedef __attribute__((address_space(3))) v4i* lv4i_p;

#if defined(__has_builtin)
#if __has_builtin(__builtin_amdgcn_global_load_async_to_lds_b128)
#define HAVE_ASYNC_LDS 1
#endif
#endif

#ifndef HAVE_ASYNC_LDS
#define HAVE_ASYNC_LDS 0
#endif

__device__ __forceinline__ void wait_async_all() {
#if HAVE_ASYNC_LDS
#if __has_builtin(__builtin_amdgcn_s_wait_asynccnt)
    __builtin_amdgcn_s_wait_asynccnt(0);
#else
    asm volatile("s_wait_asynccnt 0x0" ::: "memory");
#endif
#endif
}

union V16 { v16bf v; v8bf h[2]; };

__device__ __forceinline__ __bf16 f2bf(float f) {
    uint32_t u = __builtin_bit_cast(uint32_t, f);
    uint32_t r = (u + 0x7FFFu + ((u >> 16) & 1u)) >> 16;   // round-to-nearest-even
    uint16_t s = (uint16_t)r;
    return __builtin_bit_cast(__bf16, s);
}

__device__ __forceinline__ v4bf f2bf4(v4f f) {
    v4bf r;
    r[0] = f2bf(f.x); r[1] = f2bf(f.y); r[2] = f2bf(f.z); r[3] = f2bf(f.w);
    return r;
}

// ---------------------------------------------------------------------------
// GEMM: C[M,N] = A[M,K] * W[N,K]^T + bias[N]
//   A: f32 or bf16 (template), row-major, stride K
//   W: f32, row-major N x K (so B[k,n] = W[n,k], contiguous in k)
//   C: bf16 or f32 (template), row-major, stride N
// Block: 256 threads (8 waves), tile 128x128, K-step 64, software-pipelined
// global loads (f32->bf16 conversion needs the VGPR trip, so no async here).
// ---------------------------------------------------------------------------
template<bool A_BF16, bool OUT_BF16>
__global__ __launch_bounds__(256)
void gemm_bf16_kernel(const void* __restrict__ Aptr,
                      const float* __restrict__ W,
                      const float* __restrict__ bias,
                      void* __restrict__ Cptr,
                      int M, int N, int K)
{
    __shared__ __bf16 As[128][72];   // [row][k], padded: conflict-free
    __shared__ __bf16 Bs[128][72];   // [n][k]

    const size_t m0 = (size_t)blockIdx.y * 128;
    const size_t n0 = (size_t)blockIdx.x * 128;

    const int tid  = threadIdx.x;
    const int wid  = tid >> 5;
    const int lane = tid & 31;
    const int hf   = lane >> 4;      // half-wave: 0 or 1
    const int l16  = lane & 15;
    const int wm   = wid & 3;        // 0..3 -> 32-row strip
    const int wn   = wid >> 2;       // 0..1 -> 64-col strip

    v8f acc[2][4];
    #pragma unroll
    for (int r = 0; r < 2; ++r)
        for (int c = 0; c < 4; ++c)
            for (int e = 0; e < 8; ++e) acc[r][c][e] = 0.0f;

    const int tcol = (tid & 15) * 4;  // k offset 0..60
    const int trow = tid >> 4;        // 0..15

    // Staged (registers) next K-slice, already converted to bf16
    v4bf ra[8], rb[8];

    auto stage = [&](int kk) {
        #pragma unroll
        for (int p = 0; p < 8; ++p) {
            int row = trow + p * 16;
            if constexpr (A_BF16) {
                const __bf16* src = (const __bf16*)Aptr + (m0 + row) * (size_t)K + kk + tcol;
                ra[p] = *(const v4bf*)src;
            } else {
                const float* src = (const float*)Aptr + (m0 + row) * (size_t)K + kk + tcol;
                ra[p] = f2bf4(*(const v4f*)src);
            }
            const float* srcB = W + (n0 + row) * (size_t)K + kk + tcol;
            rb[p] = f2bf4(*(const v4f*)srcB);
        }
    };

    stage(0);

    for (int kk = 0; kk < K; kk += 64) {
        // Commit staged slice to LDS
        #pragma unroll
        for (int p = 0; p < 8; ++p) {
            int row = trow + p * 16;
            *(v4bf*)&As[row][tcol] = ra[p];
            *(v4bf*)&Bs[row][tcol] = rb[p];
        }
        __syncthreads();

        if (kk + 64 < K) stage(kk + 64);   // overlap global loads with WMMAs

        #pragma unroll
        for (int ks = 0; ks < 2; ++ks) {   // two 32-deep sub-steps
            V16 af[2];
            #pragma unroll
            for (int r = 0; r < 2; ++r) {
                int row = wm * 32 + r * 16 + l16;
                int k0 = ks * 32 + hf * 8;
                af[r].h[0] = *(const v8bf*)&As[row][k0];
                af[r].h[1] = *(const v8bf*)&As[row][k0 + 16];
            }
            #pragma unroll
            for (int c = 0; c < 4; ++c) {
                int n = wn * 64 + c * 16 + l16;
                V16 bf;
                bf.h[0] = *(const v8bf*)&Bs[n][ks * 32 + 8 * hf];
                bf.h[1] = *(const v8bf*)&Bs[n][ks * 32 + 8 * hf + 16];
                #pragma unroll
                for (int r = 0; r < 2; ++r)
                    acc[r][c] = __builtin_amdgcn_wmma_f32_16x16x32_bf16(
                        false, af[r].v, false, bf.v, (short)0, acc[r][c], false, false);
            }
        }
        __syncthreads();
    }

    // Epilogue: C element e -> row = e + 8*hf, col = l16 (per 16x16 subtile)
    #pragma unroll
    for (int c = 0; c < 4; ++c) {
        size_t n = n0 + wn * 64 + c * 16 + l16;
        float bv = bias[n];
        #pragma unroll
        for (int r = 0; r < 2; ++r) {
            size_t mbase = m0 + wm * 32 + r * 16 + hf * 8;
            #pragma unroll
            for (int e = 0; e < 8; ++e) {
                float v = acc[r][c][e] + bv;
                size_t idx = (mbase + e) * (size_t)N + n;
                if constexpr (OUT_BF16) ((__bf16*)Cptr)[idx] = f2bf(v);
                else                    ((float*)Cptr)[idx]  = v;
            }
        }
    }
}

// ---------------------------------------------------------------------------
// Flash attention over the reshaped h buffer.
// h viewed as 64 slabs of [2048 rows x 192 cols] bf16; per row: Q[0:64],
// K[64:128], V[128:192].  scale = 1/sqrt(64).
// Grid: (16 q-blocks, 64 slabs); 256 threads = 8 waves; wave owns 16 queries.
// Key tile = 64 keys/iteration, double-buffered in LDS -> ONE barrier per
// tile.  K tiles stream via GLOBAL_LOAD_ASYNC_TO_LDS (ASYNCcnt) when the
// toolchain exposes the builtin; V tiles (which need a 16-bit transpose) go
// through registers and are committed after the PV WMMAs so their loadcnt
// wait hides under compute.
// ---------------------------------------------------------------------------
__global__ __launch_bounds__(256)
void attn_kernel(const __bf16* __restrict__ hbf, __bf16* __restrict__ obuf)
{
    __shared__ __bf16 Ks[2][64][72];    // [buf][key_local][d]   (padded)
    __shared__ __bf16 Vst[2][64][72];   // [buf][d][key_local]   (transposed)
    __shared__ __bf16 Ps[8][16][72];    // per-wave P tile [m][key_local]

    const int g    = blockIdx.y;                  // slab = b*16 + head
    const int tid  = threadIdx.x;
    const int wid  = tid >> 5;
    const int lane = tid & 31;
    const int hf   = lane >> 4;
    const int l16  = lane & 15;
    const int qbase = blockIdx.x * 128 + wid * 16;

    const __bf16* slab = hbf + (size_t)g * (2048 * 192);

    // Q fragments for the two d-steps (A 16x32 layout), loaded once.
    V16 qf[2];
    {
        const __bf16* qrow = slab + (size_t)(qbase + l16) * 192;
        #pragma unroll
        for (int ks = 0; ks < 2; ++ks) {
            int d0 = ks * 32 + hf * 8;
            qf[ks].h[0] = *(const v8bf*)(qrow + d0);
            qf[ks].h[1] = *(const v8bf*)(qrow + d0 + 16);
        }
    }

    float mrow[8], lrow[8];
    #pragma unroll
    for (int r = 0; r < 8; ++r) { mrow[r] = -3.0e38f; lrow[r] = 0.0f; }
    v8f oacc[4];
    #pragma unroll
    for (int c = 0; c < 4; ++c)
        for (int e = 0; e < 8; ++e) oacc[c][e] = 0.0f;

    const int ldk = tid >> 3;          // 0..31: key within half-tile
    const int ldd = (tid & 7) * 8;     // 0..56: d offset

    v8bf vreg[2];                      // staged V (2 passes of 32 keys)
#if !HAVE_ASYNC_LDS
    v8bf kreg[2];                      // fallback: staged K
#endif

    auto stageK = [&](int kt, int buf) {
        #pragma unroll
        for (int ps = 0; ps < 2; ++ps) {
            int key = ps * 32 + ldk;
            const __bf16* src = slab + (size_t)(kt * 64 + key) * 192 + 64 + ldd;
#if HAVE_ASYNC_LDS
            __builtin_amdgcn_global_load_async_to_lds_b128(
                (gv4i_p)src, (lv4i_p)&Ks[buf][key][ldd], 0, 0);
#else
            kreg[ps] = *(const v8bf*)src;
#endif
        }
    };
#if !HAVE_ASYNC_LDS
    auto commitK = [&](int buf) {
        #pragma unroll
        for (int ps = 0; ps < 2; ++ps)
            *(v8bf*)&Ks[buf][ps * 32 + ldk][ldd] = kreg[ps];
    };
#endif
    auto stageV = [&](int kt) {
        #pragma unroll
        for (int ps = 0; ps < 2; ++ps)
            vreg[ps] = *(const v8bf*)(slab + (size_t)(kt * 64 + ps * 32 + ldk) * 192 + 128 + ldd);
    };
    auto commitV = [&](int buf) {
        #pragma unroll
        for (int ps = 0; ps < 2; ++ps) {
            int key = ps * 32 + ldk;
            #pragma unroll
            for (int i = 0; i < 8; ++i) Vst[buf][ldd + i][key] = vreg[ps][i];
        }
    };

    // Prologue: fill buffer 0
    stageK(0, 0);
    stageV(0);
    commitV(0);
#if !HAVE_ASYNC_LDS
    commitK(0);
#endif
    wait_async_all();
    __syncthreads();

    for (int kt = 0; kt < 32; ++kt) {  // 32 tiles x 64 keys = 2048
        const int cur = kt & 1, nxt = cur ^ 1;
        const bool more = (kt + 1) < 32;

        if (more) {                    // start streaming next tile
            stageK(kt + 1, nxt);       // async: lands directly in LDS
            stageV(kt + 1);            // sync loads into regs
        }

        // S = Q @ K^T for four 16-key subtiles
        v8f sc[4];
        #pragma unroll
        for (int t2 = 0; t2 < 4; ++t2) {
            #pragma unroll
            for (int e = 0; e < 8; ++e) sc[t2][e] = 0.0f;
            int keyl = t2 * 16 + l16;
            #pragma unroll
            for (int ks = 0; ks < 2; ++ks) {
                V16 kf;
                kf.h[0] = *(const v8bf*)&Ks[cur][keyl][ks * 32 + 8 * hf];
                kf.h[1] = *(const v8bf*)&Ks[cur][keyl][ks * 32 + 8 * hf + 16];
                sc[t2] = __builtin_amdgcn_wmma_f32_16x16x32_bf16(
                    false, qf[ks].v, false, kf.v, (short)0, sc[t2], false, false);
            }
        }

        // Online softmax: row r of subtile lives in element r of one half-wave
        float scale_o[8];
        #pragma unroll
        for (int r = 0; r < 8; ++r) {
            float s0 = sc[0][r] * 0.125f;
            float s1 = sc[1][r] * 0.125f;
            float s2 = sc[2][r] * 0.125f;
            float s3 = sc[3][r] * 0.125f;
            float mx = fmaxf(fmaxf(s0, s1), fmaxf(s2, s3));
            #pragma unroll
            for (int off = 1; off < 16; off <<= 1)
                mx = fmaxf(mx, __shfl_xor(mx, off, 32));
            float mnew = fmaxf(mrow[r], mx);
            float so = __expf(mrow[r] - mnew);
            float p0 = __expf(s0 - mnew);
            float p1 = __expf(s1 - mnew);
            float p2 = __expf(s2 - mnew);
            float p3 = __expf(s3 - mnew);
            float rs = (p0 + p1) + (p2 + p3);
            #pragma unroll
            for (int off = 1; off < 16; off <<= 1)
                rs += __shfl_xor(rs, off, 32);
            lrow[r] = lrow[r] * so + rs;
            mrow[r] = mnew;
            scale_o[r] = so;
            sc[0][r] = p0; sc[1][r] = p1; sc[2][r] = p2; sc[3][r] = p3;
        }

        // Re-layout P (16x64) into A-fragments via wave-private LDS
        #pragma unroll
        for (int t2 = 0; t2 < 4; ++t2)
            #pragma unroll
            for (int r = 0; r < 8; ++r)
                Ps[wid][r + 8 * hf][t2 * 16 + l16] = f2bf(sc[t2][r]);

        V16 pf[2];
        #pragma unroll
        for (int pk = 0; pk < 2; ++pk) {
            pf[pk].h[0] = *(const v8bf*)&Ps[wid][l16][pk * 32 + hf * 8];
            pf[pk].h[1] = *(const v8bf*)&Ps[wid][l16][pk * 32 + hf * 8 + 16];
        }

        // Rescale running O, then O += P @ V  (two 32-deep sub-steps)
        #pragma unroll
        for (int c = 0; c < 4; ++c)
            #pragma unroll
            for (int e = 0; e < 8; ++e) oacc[c][e] *= scale_o[e];
        #pragma unroll
        for (int c = 0; c < 4; ++c) {
            #pragma unroll
            for (int pk = 0; pk < 2; ++pk) {
                V16 vf;
                vf.h[0] = *(const v8bf*)&Vst[cur][c * 16 + l16][pk * 32 + 8 * hf];
                vf.h[1] = *(const v8bf*)&Vst[cur][c * 16 + l16][pk * 32 + 8 * hf + 16];
                oacc[c] = __builtin_amdgcn_wmma_f32_16x16x32_bf16(
                    false, pf[pk].v, false, vf.v, (short)0, oacc[c], false, false);
            }
        }

        // Commit next V tile (loadcnt wait hidden under the WMMAs above)
        if (more) {
            commitV(nxt);
#if !HAVE_ASYNC_LDS
            commitK(nxt);
#endif
        }
        wait_async_all();
        __syncthreads();
    }

    // Finalize: O /= l ; write transposed layout obuf[(b*2048+q)*1024 + head*64 + d]
    const int b = g >> 4, head = g & 15;
    #pragma unroll
    for (int c = 0; c < 4; ++c) {
        int d = head * 64 + c * 16 + l16;
        #pragma unroll
        for (int e = 0; e < 8; ++e) {
            int q = qbase + e + 8 * hf;
            float v = oacc[c][e] / lrow[e];
            obuf[((size_t)(b * 2048 + q)) * 1024 + d] = f2bf(v);
        }
    }
}

// ---------------------------------------------------------------------------
extern "C" void kernel_launch(void* const* d_in, const int* in_sizes, int n_in,
                              void* d_out, int out_size, void* d_ws, size_t ws_size,
                              hipStream_t stream)
{
    (void)in_sizes; (void)n_in; (void)out_size; (void)ws_size;

    const float* x    = (const float*)d_in[0];   // (4,2048,1024)
    const float* Wqkv = (const float*)d_in[1];   // (3072,1024)
    const float* bqkv = (const float*)d_in[2];   // (3072,)
    const float* Wo   = (const float*)d_in[3];   // (1024,1024)
    const float* bo   = (const float*)d_in[4];   // (1024,)
    float* out = (float*)d_out;                  // (4,2048,1024) f32

    __bf16* hbf  = (__bf16*)d_ws;                     // 8192x3072 bf16 (48 MB)
    __bf16* obuf = hbf + (size_t)8192 * 3072;         // 8192x1024 bf16 (16 MB)

    dim3 blk(256);

    // h = x @ Wqkv^T + b_qkv   (M=8192, N=3072, K=1024), bf16 out
    gemm_bf16_kernel<false, true><<<dim3(3072 / 128, 8192 / 128), blk, 0, stream>>>(
        x, Wqkv, bqkv, hbf, 8192, 3072, 1024);

    // attention over 64 slabs, flash-style, writes pre-transposed O
    attn_kernel<<<dim3(2048 / 128, 64), blk, 0, stream>>>(hbf, obuf);

    // out = O @ Wo^T + b_o     (M=8192, N=1024, K=1024), f32 out
    gemm_bf16_kernel<true, false><<<dim3(1024 / 128, 8192 / 128), blk, 0, stream>>>(
        obuf, Wo, bo, out, 8192, 1024, 1024);
}
